// VanillaLSTMCell_80350248173766
// MI455X (gfx1250) — matrix operand order repaired and checked by reading, not compiled, not run on hardware
//
#include <hip/hip_runtime.h>
#include <cstdint>
#include <cstddef>

// ---------------------------------------------------------------------------
// VanillaLSTMCell for MI455X (gfx1250, wave32, WMMA)
//   B=8192, I=H=1024.  z = x@WxT + h@WhT + b  -> gates -> h_next, c_next
//   Phase 1: pack weights transposed to bf16 [K][4H], x/h to bf16, fuse biases
//   Phase 2: fused WMMA GEMM (bf16 in, f32 acc) + LSTM epilogue
//   v2: K-stage = 64 (fewer barriers), batched B-fragment loads so the 8/16
//       WMMAs per stage stream back-to-back instead of stalling on dscnt.
// ---------------------------------------------------------------------------

#define B_DIM 8192
#define H_DIM 1024
#define FOURH 4096

typedef __attribute__((ext_vector_type(16))) __bf16 v16bf;
typedef __attribute__((ext_vector_type(8)))  float  v8f;

struct U32x8 { uint4 a, b; };

__device__ __forceinline__ v16bf make_frag(uint4 lo, uint4 hi) {
  U32x8 t; t.a = lo; t.b = hi;
  return __builtin_bit_cast(v16bf, t);
}

__device__ __forceinline__ unsigned short f2bf(float f) {
  unsigned u = __builtin_bit_cast(unsigned, f);
  u += 0x7fffu + ((u >> 16) & 1u);           // round-to-nearest-even
  return (unsigned short)(u >> 16);
}

__device__ __forceinline__ float sigmoidf_(float x) {
  return 1.0f / (1.0f + __expf(-x));
}

// ---------------------------------------------------------------------------
// Pack kernels
// ---------------------------------------------------------------------------

// f32 -> bf16, 4 elements/thread (b128 in, b64 out)
__global__ __launch_bounds__(256) void cvt_bf16x4_kernel(
    const float* __restrict__ in, unsigned short* __restrict__ out, int n4) {
  int i = blockIdx.x * blockDim.x + threadIdx.x;
  if (i >= n4) return;
  float4 f = ((const float4*)in)[i];
  ushort4 r;
  r.x = f2bf(f.x); r.y = f2bf(f.y); r.z = f2bf(f.z); r.w = f2bf(f.w);
  ((ushort4*)out)[i] = r;
}

// W [1024 n][1024 k] f32 (row-major) -> WT[k][4096] bf16 at column colOff
__global__ __launch_bounds__(256) void transpose_pack_kernel(
    const float* __restrict__ src, unsigned short* __restrict__ dst, int colOff) {
  __shared__ float tile[32][33];
  const int tx = threadIdx.x;        // 0..31
  const int ty = threadIdx.y;        // 0..7
  const int n0 = blockIdx.x * 32;
  const int k0 = blockIdx.y * 32;
#pragma unroll
  for (int r = 0; r < 4; ++r)
    tile[ty + 8 * r][tx] = src[(size_t)(n0 + ty + 8 * r) * 1024 + (k0 + tx)];
  __syncthreads();
#pragma unroll
  for (int r = 0; r < 4; ++r) {
    int k = k0 + ty + 8 * r;
    dst[(size_t)k * FOURH + colOff + n0 + tx] = f2bf(tile[tx][ty + 8 * r]);
  }
}

__global__ __launch_bounds__(256) void bias_add_kernel(
    const float* __restrict__ a, const float* __restrict__ b, float* __restrict__ dst) {
  int i = blockIdx.x * blockDim.x + threadIdx.x;
  if (i < 1024) dst[i] = a[i] + b[i];
}

// ---------------------------------------------------------------------------
// Fused WMMA GEMM + LSTM epilogue
//   grid: (H/64, B/64), block: 256 (8 waves = 4 Mrows x 2 Ncols)
//   WG tile: 64 M x 64 N per gate x 4 gates
//   Weights staged to LDS via async global->LDS b128, double buffered,
//   K-stage = 64 (32 KB/stage).  LDS row pitch 528 B -> conflict-free
//   ds_load_u16 fragment reads.
// ---------------------------------------------------------------------------

#define KSTAGE    64
#define NSTAGE    32                     // 2048 total K / 64
#define LDS_PITCH 528
#define LDS_BUFB  (KSTAGE * LDS_PITCH)   // 33792 B per buffer

__global__ __launch_bounds__(256) void lstm_wmma_kernel(
    const unsigned short* __restrict__ xb,
    const unsigned short* __restrict__ hb,
    const unsigned short* __restrict__ WxT,
    const unsigned short* __restrict__ WhT,
    const float* __restrict__ bias,
    const float* __restrict__ c_in,
    float* __restrict__ out) {
  __shared__ __align__(16) unsigned char ldsB[2][LDS_BUFB];

  const int t    = threadIdx.x;
  const int lane = t & 31;
  const int w    = t >> 5;
  const int wr   = w >> 1;     // 0..3  (M row of wave)
  const int wc   = w & 1;      // 0..1  (N half of wave)
  const int N0   = blockIdx.x * 64;
  const int M0   = blockIdx.y * 64;

  const unsigned ldsBase = (unsigned)(unsigned long long)(uintptr_t)(&ldsB[0][0]);

  v8f acc[4][2];
#pragma unroll
  for (int g = 0; g < 4; ++g)
#pragma unroll
    for (int j = 0; j < 2; ++j)
      acc[g][j] = (v8f){0.f, 0.f, 0.f, 0.f, 0.f, 0.f, 0.f, 0.f};

  // issue one stage's weight tile (64 K x 256 N bf16 = 32KB) into LDS buffer
  auto issue = [&](int s) {
    const unsigned short* Bb = (s < 16) ? WxT : WhT;
    const unsigned long long base = (unsigned long long)(uintptr_t)Bb;
    const int k0 = (s & 15) * KSTAGE;
    const unsigned buf = ldsBase + (unsigned)((s & 1) * LDS_BUFB);
#pragma unroll
    for (int i = 0; i < 8; ++i) {
      int id  = i * 256 + t;
      int k   = id >> 5;        // 0..63  K row
      int sub = id & 31;
      int g   = sub >> 3;       // gate 0..3
      int ch  = sub & 7;        // 16B chunk within 128B gate segment
      unsigned goff = (unsigned)((((k0 + k) << 12) + (g << 10) + N0 + (ch << 3)) * 2);
      unsigned loff = buf + (unsigned)(k * LDS_PITCH + g * 128 + ch * 16);
      asm volatile("global_load_async_to_lds_b128 %0, %1, %2"
                   :: "v"(loff), "v"(goff), "s"(base)
                   : "memory");
    }
  };

  const int mRow = M0 + wr * 16 + (lane & 15);
  const int kSub = (lane >> 4) << 3;     // 0 or 8

  issue(0);
  for (int s = 0; s < NSTAGE; ++s) {
    const int cur = s & 1;
    if (s + 1 < NSTAGE) {
      issue(s + 1);
      asm volatile("s_wait_asynccnt 0x8" ::: "memory");  // stage s's 8 copies done
    } else {
      asm volatile("s_wait_asynccnt 0x0" ::: "memory");
    }
    __syncthreads();

    // Both A fragments for this stage up-front (4x global b128, long latency)
    const unsigned short* Ab = (s < 16) ? xb : hb;
    const int k0 = (s & 15) * KSTAGE;
    const unsigned short* ar = Ab + (size_t)mRow * 1024 + k0 + kSub;
    v16bf afrag[2];
    afrag[0] = make_frag(*(const uint4*)ar,        *(const uint4*)(ar + 16));
    afrag[1] = make_frag(*(const uint4*)(ar + 32), *(const uint4*)(ar + 48));

#pragma unroll
    for (int half = 0; half < 2; ++half) {
      // Batch-load all 8 B fragments (16x ds_load_u16 each, column n = lane%16,
      // k in table order), then stream 8 WMMAs back-to-back.
      const unsigned char* bbase =
          &ldsB[cur][(half * 32 + kSub) * LDS_PITCH + wc * 64 + (lane & 15) * 2];
      v16bf bfrag[8];
#pragma unroll
      for (int gj = 0; gj < 8; ++gj) {
        const unsigned char* p = bbase + (gj >> 1) * 128 + (gj & 1) * 32;
        v16bf f;
#pragma unroll
        for (int e = 0; e < 16; ++e)
          f[e] = *(const __bf16*)(p + ((e & 7) + ((e >> 3) << 4)) * LDS_PITCH);
        bfrag[gj] = f;
      }
#pragma unroll
      for (int gj = 0; gj < 8; ++gj) {
        acc[gj >> 1][gj & 1] = __builtin_amdgcn_wmma_f32_16x16x32_bf16(
            false, afrag[half], false, bfrag[gj], (short)0,
            acc[gj >> 1][gj & 1], false, false);
      }
    }
    __syncthreads();   // all reads of buffer `cur` done before it is re-filled
  }

  // ---- LSTM epilogue, fully in registers ----
  const int mB = M0 + wr * 16 + ((lane >> 4) << 3);
  const int nB = N0 + wc * 32 + (lane & 15);
  float* __restrict__ outH = out;
  float* __restrict__ outC = out + (size_t)B_DIM * H_DIM;

#pragma unroll
  for (int j = 0; j < 2; ++j) {
    const int n = nB + j * 16;
    const float bi = bias[n];
    const float bf = bias[1024 + n];
    const float bo = bias[2048 + n];
    const float bg = bias[3072 + n];
#pragma unroll
    for (int i = 0; i < 8; ++i) {
      const int m = mB + i;
      const size_t idx = (size_t)m * H_DIM + n;
      float zi = acc[0][j][i] + bi;
      float zf = acc[1][j][i] + bf;
      float zo = acc[2][j][i] + bo;
      float zg = acc[3][j][i] + bg;
      float it = sigmoidf_(zi);
      float ft = sigmoidf_(zf);
      float ot = sigmoidf_(zo);
      float gt = tanhf(zg);
      float cn = ft * c_in[idx] + it * gt;
      outC[idx] = cn;
      outH[idx] = ot * tanhf(cn);
    }
  }
}

// ---------------------------------------------------------------------------
// Launcher
// ---------------------------------------------------------------------------
extern "C" void kernel_launch(void* const* d_in, const int* in_sizes, int n_in,
                              void* d_out, int out_size, void* d_ws, size_t ws_size,
                              hipStream_t stream) {
  (void)in_sizes; (void)n_in; (void)out_size; (void)ws_size;

  const float* x = (const float*)d_in[0];
  const float* h = (const float*)d_in[1];
  const float* c = (const float*)d_in[2];
  // dict order: Wxi, Whi, Wxf, Whf, Wxo, Who, Wxg, Whg -- each _w then _b
  const float* Wx_w[4] = { (const float*)d_in[3],  (const float*)d_in[7],
                           (const float*)d_in[11], (const float*)d_in[15] };
  const float* Wx_b[4] = { (const float*)d_in[4],  (const float*)d_in[8],
                           (const float*)d_in[12], (const float*)d_in[16] };
  const float* Wh_w[4] = { (const float*)d_in[5],  (const float*)d_in[9],
                           (const float*)d_in[13], (const float*)d_in[17] };
  const float* Wh_b[4] = { (const float*)d_in[6],  (const float*)d_in[10],
                           (const float*)d_in[14], (const float*)d_in[18] };

  // workspace layout (bf16 as ushort): xb, hb, WxT, WhT, bias  (~48 MB)
  unsigned short* xb  = (unsigned short*)d_ws;
  unsigned short* hb  = xb  + (size_t)B_DIM * 1024;
  unsigned short* WxT = hb  + (size_t)B_DIM * 1024;
  unsigned short* WhT = WxT + (size_t)1024 * FOURH;
  float*          bia = (float*)(WhT + (size_t)1024 * FOURH);

  // Phase 1: pack
  const int n4 = (B_DIM * 1024) / 4;            // 2,097,152
  cvt_bf16x4_kernel<<<n4 / 256, 256, 0, stream>>>(x, xb, n4);
  cvt_bf16x4_kernel<<<n4 / 256, 256, 0, stream>>>(h, hb, n4);
  for (int g = 0; g < 4; ++g) {
    transpose_pack_kernel<<<dim3(32, 32), dim3(32, 8), 0, stream>>>(Wx_w[g], WxT, g * 1024);
    transpose_pack_kernel<<<dim3(32, 32), dim3(32, 8), 0, stream>>>(Wh_w[g], WhT, g * 1024);
    bias_add_kernel<<<4, 256, 0, stream>>>(Wx_b[g], Wh_b[g], bia + g * 1024);
  }

  // Phase 2: fused WMMA GEMM + LSTM
  lstm_wmma_kernel<<<dim3(H_DIM / 64, B_DIM / 64), 256, 0, stream>>>(
      xb, hb, WxT, WhT, bia, c, (float*)d_out);
}
